// PureGNN_54202487275747
// MI455X (gfx1250) — matrix operand order repaired and checked by reading, compiled, not run on hardware
//
#include <hip/hip_runtime.h>
#include <hip/hip_bf16.h>

// CDNA5 / gfx1250, wave32 WMMA GNN implementation.
// N=50000 nodes, D=H=128, E=600000 edges, L=3 layers.

typedef __attribute__((ext_vector_type(16))) _Float16 v16h;
typedef __attribute__((ext_vector_type(8)))  _Float16 v8h;
typedef __attribute__((ext_vector_type(8)))  float    v8f;

#define HDIM 128
#define KDIM 256   // 2*H for GNN layers

__device__ __forceinline__ void atomic_add_f32(float* p, float v) {
    __hip_atomic_fetch_add(p, v, __ATOMIC_RELAXED, __HIP_MEMORY_SCOPE_AGENT);
}

// ---------------------------------------------------------------------------
// Pack weights: transpose to Wt[n][k] in f16 so a WMMA B-fragment per lane is
// 32 contiguous bytes. Also zero the column-sum accumulator.
// ---------------------------------------------------------------------------
__global__ void pack_kernel(const float* __restrict__ embW,   // [128,128] (k,n)
                            const float* __restrict__ gnnW,   // [3,256,128] (l,k,n)
                            _Float16* __restrict__ embT,      // [128,128] (n,k)
                            _Float16* __restrict__ gnnT,      // [3,128,256] (l,n,k)
                            float* __restrict__ colSum) {
    int i = blockIdx.x * blockDim.x + threadIdx.x;
    int stride = gridDim.x * blockDim.x;
    for (int idx = i; idx < 128 * 128; idx += stride) {
        int n = idx >> 7, k = idx & 127;
        embT[idx] = (_Float16)embW[k * 128 + n];
    }
    for (int idx = i; idx < 3 * 128 * 256; idx += stride) {
        int l = idx / 32768;
        int rem = idx & 32767;
        int n = rem >> 8, k = rem & 255;
        gnnT[idx] = (_Float16)gnnW[l * 32768 + k * 128 + n];
    }
    if (i < 128) colSum[i] = 0.0f;
}

// ---------------------------------------------------------------------------
// Embed: h = relu(X @ embed_W + b).  64 nodes per block, 4 waves, each wave
// computes a 16x128 output stripe with v_wmma_f32_16x16x32_f16 (K=128).
// ---------------------------------------------------------------------------
__global__ void embed_kernel(const float* __restrict__ X,
                             const _Float16* __restrict__ Wt,   // [128][128] (n,k)
                             const float* __restrict__ bias,
                             float* __restrict__ hOut, int nNodes) {
    __shared__ _Float16 sA[64 * 136];   // 136 = 128 + 8 pad halves (16B-aligned rows)
    const int tid = threadIdx.x;
    const int rowBase = blockIdx.x * 64;

    // Load 64 rows of X (f32 -> f16) into LDS; 2 threads per row.
    {
        int r = tid >> 1, cs = (tid & 1) * 64;
        int node = rowBase + r;
        _Float16* dp = &sA[r * 136 + cs];
        if (node < nNodes) {
            const float* sp = X + (size_t)node * HDIM + cs;
            #pragma unroll
            for (int c = 0; c < 64; c += 4) {
                float4 f = *(const float4*)(sp + c);
                dp[c + 0] = (_Float16)f.x; dp[c + 1] = (_Float16)f.y;
                dp[c + 2] = (_Float16)f.z; dp[c + 3] = (_Float16)f.w;
            }
        } else {
            #pragma unroll
            for (int c = 0; c < 64; ++c) dp[c] = (_Float16)0.0f;
        }
    }
    __syncthreads();

    const int wave = tid >> 5, lane = tid & 31;
    const int lrow = lane & 15, hs = lane >> 4;
    const int aRow = wave * 16 + lrow;

    v8f zero8 = {0.f, 0.f, 0.f, 0.f, 0.f, 0.f, 0.f, 0.f};
    v8f c[8];
    #pragma unroll
    for (int i = 0; i < 8; ++i) c[i] = zero8;

    #pragma unroll
    for (int kt = 0; kt < 4; ++kt) {
        v16h a;
        ((v8h*)&a)[0] = *(const v8h*)&sA[aRow * 136 + kt * 32 + hs * 16];
        ((v8h*)&a)[1] = *(const v8h*)&sA[aRow * 136 + kt * 32 + hs * 16 + 8];
        #pragma unroll
        for (int nt = 0; nt < 8; ++nt) {
            const v8h* wp = (const v8h*)(Wt + (size_t)(nt * 16 + lrow) * 128 + kt * 32 + hs * 16);
            v16h b;
            ((v8h*)&b)[0] = wp[0];
            ((v8h*)&b)[1] = wp[1];
            c[nt] = __builtin_amdgcn_wmma_f32_16x16x32_f16(
                false, a, false, b, (short)0, c[nt], false, false);
        }
    }

    #pragma unroll
    for (int nt = 0; nt < 8; ++nt) {
        int n = nt * 16 + lrow;
        float bn = bias[n];
        #pragma unroll
        for (int v = 0; v < 8; ++v) {
            int m = rowBase + wave * 16 + v + 8 * hs;
            if (m < nNodes) {
                hOut[(size_t)m * HDIM + n] = fmaxf(c[nt][v] + bn, 0.0f);
            }
        }
    }
}

// ---------------------------------------------------------------------------
// Prep per layer: h_next = h_cur (scatter-add target init) and hHalf = f16(h).
// ---------------------------------------------------------------------------
__global__ void prep_kernel(const float* __restrict__ hCur,
                            float* __restrict__ hNextInit,
                            _Float16* __restrict__ hHalf, int total) {
    int i = blockIdx.x * blockDim.x + threadIdx.x;
    int stride = gridDim.x * blockDim.x;
    for (; i < total; i += stride) {
        float v = hCur[i];
        hNextInit[i] = v;
        hHalf[i] = (_Float16)v;
    }
}

// ---------------------------------------------------------------------------
// GNN layer: per block, 128 edges (4 waves x 32-edge M-tiles). Gather
// concat(h[src],h[dst]) in f16 to LDS; each wave holds TWO 16x128 accumulator
// stripes so every B fragment (1KB/wave from cache) feeds two WMMAs,
// halving the dominant cache-read traffic per v_wmma. Bias+relu, then
// scatter-add into hNext (== segment_sum by dst) with native f32 atomics.
// ---------------------------------------------------------------------------
__global__ void gnn_layer_kernel(const _Float16* __restrict__ hHalf,
                                 const int* __restrict__ srcArr,
                                 const int* __restrict__ dstArr,
                                 const _Float16* __restrict__ Wt,  // [128][256] (n,k)
                                 const float* __restrict__ bias,
                                 float* __restrict__ hNext, int nEdges) {
    __shared__ _Float16 sA[128 * 264];  // 264 = 256 + 8 pad halves; 66KB
    const int tid = threadIdx.x;        // 128 threads, 4 waves
    const int eBase = blockIdx.x * 128;

    // Gather: 256 row-halves (128 edges x {src,dst}); 2 per thread.
    #pragma unroll
    for (int j = tid; j < 256; j += 128) {
        int r = j >> 1, half = j & 1;
        int e = eBase + r;
        int node = 0;
        if (e < nEdges) node = half ? dstArr[e] : srcArr[e];
        const v8h* sp = (const v8h*)(hHalf + (size_t)node * HDIM);
        v8h* dp = (v8h*)&sA[(size_t)r * 264 + half * HDIM];
        #pragma unroll
        for (int cc = 0; cc < 16; ++cc) dp[cc] = sp[cc];
    }
    __syncthreads();

    const int wave = tid >> 5, lane = tid & 31;
    const int lrow = lane & 15, hs = lane >> 4;
    const int aRow0 = wave * 32 + lrow;   // first 16-edge subtile
    const int aRow1 = aRow0 + 16;         // second 16-edge subtile

    v8f zero8 = {0.f, 0.f, 0.f, 0.f, 0.f, 0.f, 0.f, 0.f};
    v8f c0[8], c1[8];
    #pragma unroll
    for (int i = 0; i < 8; ++i) { c0[i] = zero8; c1[i] = zero8; }

    #pragma unroll
    for (int kt = 0; kt < 8; ++kt) {
        v16h a0, a1;
        ((v8h*)&a0)[0] = *(const v8h*)&sA[aRow0 * 264 + kt * 32 + hs * 16];
        ((v8h*)&a0)[1] = *(const v8h*)&sA[aRow0 * 264 + kt * 32 + hs * 16 + 8];
        ((v8h*)&a1)[0] = *(const v8h*)&sA[aRow1 * 264 + kt * 32 + hs * 16];
        ((v8h*)&a1)[1] = *(const v8h*)&sA[aRow1 * 264 + kt * 32 + hs * 16 + 8];
        #pragma unroll
        for (int nt = 0; nt < 8; ++nt) {
            const v8h* wp = (const v8h*)(Wt + (size_t)(nt * 16 + lrow) * KDIM + kt * 32 + hs * 16);
            v16h b;
            ((v8h*)&b)[0] = wp[0];
            ((v8h*)&b)[1] = wp[1];
            c0[nt] = __builtin_amdgcn_wmma_f32_16x16x32_f16(
                false, a0, false, b, (short)0, c0[nt], false, false);
            c1[nt] = __builtin_amdgcn_wmma_f32_16x16x32_f16(
                false, a1, false, b, (short)0, c1[nt], false, false);
        }
    }

    // Destination node per C-row (M = v + 8*hs within each 16-edge subtile).
    int dIdx0[8], dIdx1[8];
    #pragma unroll
    for (int v = 0; v < 8; ++v) {
        int e0 = eBase + wave * 32 + v + 8 * hs;
        int e1 = e0 + 16;
        dIdx0[v] = (e0 < nEdges) ? dstArr[e0] : -1;
        dIdx1[v] = (e1 < nEdges) ? dstArr[e1] : -1;
    }

    #pragma unroll
    for (int nt = 0; nt < 8; ++nt) {
        int n = nt * 16 + lrow;
        float bn = bias[n];
        #pragma unroll
        for (int v = 0; v < 8; ++v) {
            if (dIdx0[v] >= 0) {
                float val = fmaxf(c0[nt][v] + bn, 0.0f);
                atomic_add_f32(&hNext[(size_t)dIdx0[v] * HDIM + n], val);
            }
            if (dIdx1[v] >= 0) {
                float val = fmaxf(c1[nt][v] + bn, 0.0f);
                atomic_add_f32(&hNext[(size_t)dIdx1[v] * HDIM + n], val);
            }
        }
    }
}

// ---------------------------------------------------------------------------
// Column sums of final h (for the mean): each block grid-strides rows,
// thread t owns column t; one atomicAdd per (block, column).
// ---------------------------------------------------------------------------
__global__ void colsum_kernel(const float* __restrict__ h,
                              float* __restrict__ colSum, int nNodes) {
    int t = threadIdx.x;   // 128 threads
    float s = 0.0f;
    for (int n = blockIdx.x; n < nNodes; n += gridDim.x)
        s += h[(size_t)n * HDIM + t];
    atomic_add_f32(&colSum[t], s);
}

// ---------------------------------------------------------------------------
// Predictor: ctx = concat(h[0], mean(h)); p = relu(ctx@W1+b1); out = p@W2+b2.
// ---------------------------------------------------------------------------
__global__ void predict_kernel(const float* __restrict__ hFinal,
                               const float* __restrict__ colSum,
                               const float* __restrict__ W1,   // [256,128]
                               const float* __restrict__ b1,
                               const float* __restrict__ W2,   // [128]
                               const float* __restrict__ b2,
                               float* __restrict__ out, int nNodes) {
    __shared__ float sctx[256];
    __shared__ float sp[128];
    int t = threadIdx.x;   // 128 threads
    sctx[t] = hFinal[t];                        // h[0] row
    sctx[128 + t] = colSum[t] / (float)nNodes;  // mean
    __syncthreads();
    float acc = b1[t];
    #pragma unroll 4
    for (int k = 0; k < 256; ++k) acc += sctx[k] * W1[k * 128 + t];
    sp[t] = fmaxf(acc, 0.0f) * W2[t];
    __syncthreads();
    for (int s = 64; s > 0; s >>= 1) {
        if (t < s) sp[t] += sp[t + s];
        __syncthreads();
    }
    if (t == 0) out[0] = sp[0] + b2[0];
}

// ---------------------------------------------------------------------------
extern "C" void kernel_launch(void* const* d_in, const int* in_sizes, int n_in,
                              void* d_out, int out_size, void* d_ws, size_t ws_size,
                              hipStream_t stream) {
    const float* X    = (const float*)d_in[0];
    const int*   ei   = (const int*)  d_in[1];
    const float* embW = (const float*)d_in[2];
    const float* embB = (const float*)d_in[3];
    const float* gnnW = (const float*)d_in[4];
    const float* gnnB = (const float*)d_in[5];
    const float* pW1  = (const float*)d_in[6];
    const float* pB1  = (const float*)d_in[7];
    const float* pW2  = (const float*)d_in[8];
    const float* pB2  = (const float*)d_in[9];

    const int nNodes = in_sizes[0] / HDIM;      // 50000
    const int nEdges = in_sizes[1] / 2;         // 600000
    const int total  = nNodes * HDIM;

    float* out    = (float*)d_out;
    float* hFinal = out + 1;                    // returned h lives here

    // Workspace carve-up (256B aligned slots).
    auto align256 = [](size_t x) { return (x + 255) & ~(size_t)255; };
    char* ws = (char*)d_ws;
    size_t off = 0;
    _Float16* hHalf = (_Float16*)(ws + off); off = align256(off + (size_t)total * sizeof(_Float16));
    float*    hA    = (float*)   (ws + off); off = align256(off + (size_t)total * sizeof(float));
    float*    hB    = (float*)   (ws + off); off = align256(off + (size_t)total * sizeof(float));
    _Float16* embT  = (_Float16*)(ws + off); off = align256(off + (size_t)128 * 128 * sizeof(_Float16));
    _Float16* gnnT  = (_Float16*)(ws + off); off = align256(off + (size_t)3 * 128 * 256 * sizeof(_Float16));
    float*  colSum  = (float*)   (ws + off); off = align256(off + 128 * sizeof(float));

    const int* srcArr = ei;
    const int* dstArr = ei + nEdges;

    // 1) Weight packing (+ colSum zeroing).
    pack_kernel<<<128, 256, 0, stream>>>(embW, gnnW, embT, gnnT, colSum);

    // 2) Embedding GEMM -> hA.
    int embBlocks = (nNodes + 63) / 64;
    embed_kernel<<<embBlocks, 128, 0, stream>>>(X, embT, embB, hA, nNodes);

    int edgeBlocks = (nEdges + 127) / 128;

    // 3) Layer 0: hA -> hB
    prep_kernel<<<2048, 256, 0, stream>>>(hA, hB, hHalf, total);
    gnn_layer_kernel<<<edgeBlocks, 128, 0, stream>>>(hHalf, srcArr, dstArr,
                                                     gnnT, gnnB, hB, nEdges);
    // 4) Layer 1: hB -> hA
    prep_kernel<<<2048, 256, 0, stream>>>(hB, hA, hHalf, total);
    gnn_layer_kernel<<<edgeBlocks, 128, 0, stream>>>(hHalf, srcArr, dstArr,
                                                     gnnT + 32768, gnnB + 128, hA, nEdges);
    // 5) Layer 2: hA -> d_out+1 (final h written in place, atomics included)
    prep_kernel<<<2048, 256, 0, stream>>>(hA, hFinal, hHalf, total);
    gnn_layer_kernel<<<edgeBlocks, 128, 0, stream>>>(hHalf, srcArr, dstArr,
                                                     gnnT + 65536, gnnB + 256, hFinal, nEdges);

    // 6) Mean + predictor head -> d_out[0].
    colsum_kernel<<<1024, 128, 0, stream>>>(hFinal, colSum, nNodes);
    predict_kernel<<<1, 128, 0, stream>>>(hFinal, colSum, pW1, pB1, pW2, pB2,
                                          out, nNodes);
}
